// VQEmbeddingEMA_43267500540459
// MI455X (gfx1250) — matrix operand order repaired and checked by reading, not compiled, not run on hardware
//
#include <hip/hip_runtime.h>
#include <hip/hip_bf16.h>

typedef __attribute__((ext_vector_type(16))) _Float16 v16h;
typedef __attribute__((ext_vector_type(8)))  float    v8f;
typedef __attribute__((ext_vector_type(4)))  float    v4f;

#define D_DIM 256
#define TILE_STRIDE 272   // halfs; 544B row pitch -> 8-bank offset per col, breaks conflicts

// ---------------- Kernel 1: embedding f32 -> f16 + exact f32 row norms ----------------
__global__ __launch_bounds__(256)
void vq_eprep_kernel(const float* __restrict__ e, _Float16* __restrict__ eh,
                     float* __restrict__ enorm) {
    const int r = blockIdx.x;          // code row
    const int k = threadIdx.x;         // 0..255
    const float v = e[(size_t)r * D_DIM + k];
    eh[(size_t)r * D_DIM + k] = (_Float16)v;
    float s = v * v;
    #pragma unroll
    for (int m = 1; m < 32; m <<= 1) s += __shfl_xor(s, m, 32);
    __shared__ float ps[8];
    if ((threadIdx.x & 31) == 0) ps[threadIdx.x >> 5] = s;
    __syncthreads();
    if (threadIdx.x == 0) {
        float t = 0.f;
        #pragma unroll
        for (int i = 0; i < 8; ++i) t += ps[i];
        enorm[r] = t;
    }
}

// ---------------- Kernel 2: fused GEMM + argmin via WMMA f16 ----------------
// dist(n,m) = ||e_m||^2 - 2 * dot(x_n, e_m)   (||x||^2 dropped: row-constant)
// Double-buffered LDS staging (1 barrier/tile), dual WMMA accumulator chains.
__global__ __launch_bounds__(256)
void vq_argmin_kernel(const float* __restrict__ x, const _Float16* __restrict__ eh,
                      const float* __restrict__ enorm, int* __restrict__ idx_out,
                      int num_tiles) {
    __shared__ _Float16 bt[2][16 * TILE_STRIDE];   // 2 x 8704 B double-buffered B tile
    const int tid  = threadIdx.x;
    const int lane = tid & 31;
    const int wave = tid >> 5;
    const int rowBase = (blockIdx.x * 8 + wave) * 16;
    const int aRow = rowBase + (lane & 15);     // A layout: lanes 0-15 & 16-31 both M=0..15
    const int offA = (lane < 16) ? 0 : 16;      // A K offset (contiguous k = 16h + s, matches B)
    const int col  = lane & 15;                 // B/C column
    const int offB = (lane < 16) ? 0 : 16;      // B K offset (contiguous k = 16h + s, matches A)

    // Load + convert the wave's 16x256 A tile into registers (8 x v16h = 64 VGPRs)
    v16h a[8];
    const float* xr = x + (size_t)aRow * D_DIM;
    #pragma unroll
    for (int ks = 0; ks < 8; ++ks) {
        const int k0 = ks * 32;
        v16h av;
        #pragma unroll
        for (int j = 0; j < 8; ++j) av[j]     = (_Float16)xr[k0 + offA + j];
        #pragma unroll
        for (int j = 0; j < 8; ++j) av[8 + j] = (_Float16)xr[k0 + 8 + offA + j];
        a[ks] = av;
    }

    // Staging geometry: each of the 256 threads owns one 32B chunk of the 16x256h tile
    const int sc  = tid >> 4;            // 0..15 code within tile
    const int sch = (tid & 15) * 16;     // half offset 0..240
    const size_t schunk = (size_t)sc * D_DIM + sch;

    float bestV[8];
    int   bestI[8];
    #pragma unroll
    for (int r = 0; r < 8; ++r) { bestV[r] = 3.0e38f; bestI[r] = 0; }

    // Prologue: stage tile 0 into buffer 0
    {
        uint4 st = *(const uint4*)(eh + schunk);
        *(uint4*)&bt[0][sc * TILE_STRIDE + sch] = st;
    }
    __syncthreads();

    for (int t = 0; t < num_tiles; ++t) {
        // Issue global load of tile t+1 (clamped) before compute: latency hidden by WMMAs
        const int tn = (t + 1 < num_tiles) ? (t + 1) : 0;
        uint4 st = *(const uint4*)(eh + (size_t)tn * 16 * D_DIM + schunk);
        if (t + 2 < num_tiles)
            __builtin_prefetch(eh + (size_t)(t + 2) * 16 * D_DIM + schunk, 0, 3);

        const _Float16* bp = &bt[t & 1][col * TILE_STRIDE + offB];
        v8f acc0 = 0.f, acc1 = 0.f;   // two independent chains -> WMMAs can interleave
        #pragma unroll
        for (int ks = 0; ks < 8; ks += 2) {
            v16h b0 = *(const v16h*)(bp + (ks    ) * 32);
            v16h b1 = *(const v16h*)(bp + (ks + 1) * 32);
            acc0 = __builtin_amdgcn_wmma_f32_16x16x32_f16(
                false, a[ks],     false, b0, (short)0, acc0, false, false);
            acc1 = __builtin_amdgcn_wmma_f32_16x16x32_f16(
                false, a[ks + 1], false, b1, (short)0, acc1, false, false);
        }

        const int cg = t * 16 + col;
        const float en = enorm[cg];
        #pragma unroll
        for (int r = 0; r < 8; ++r) {
            const float d = en - 2.0f * (acc0[r] + acc1[r]);  // C row = r + 8*(lane>=16)
            if (d < bestV[r]) { bestV[r] = d; bestI[r] = cg; }
        }

        // Store tile t+1 into the other buffer; single barrier per iteration
        *(uint4*)&bt[(t + 1) & 1][sc * TILE_STRIDE + sch] = st;
        __syncthreads();
    }

    // Cross-lane (16-lane halves) argmin reduction; ties -> smaller index (matches argmin)
    #pragma unroll
    for (int r = 0; r < 8; ++r) {
        float v = bestV[r];
        int   i = bestI[r];
        #pragma unroll
        for (int m = 1; m < 16; m <<= 1) {
            const float ov = __shfl_xor(v, m, 32);
            const int   oi = __shfl_xor(i, m, 32);
            if (ov < v || (ov == v && oi < i)) { v = ov; i = oi; }
        }
        if ((lane & 15) == r)
            idx_out[rowBase + r + ((lane >> 4) << 3)] = i;
    }
}

// ---------------- Kernel 3: gather + straight-through + loss partials + histogram ----------------
__global__ __launch_bounds__(256)
void vq_out_kernel(const float* __restrict__ x, const float* __restrict__ e,
                   const int* __restrict__ idx, float* __restrict__ qout,
                   float* __restrict__ idxout, float* __restrict__ loss_sum,
                   float* __restrict__ nonpad_sum, unsigned int* __restrict__ counts) {
    const int lane = threadIdx.x & 31;
    const int wave = threadIdx.x >> 5;
    const int n = blockIdx.x * 8 + wave;
    const int id = idx[n];
    const float* xr = x + (size_t)n * D_DIM;
    const float* er = e + (size_t)id * D_DIM;
    float* qr = qout + (size_t)n * D_DIM;

    float msum = 0.f, asum = 0.f;
    #pragma unroll
    for (int c = 0; c < 2; ++c) {
        const int k = c * 128 + lane * 4;
        v4f xv = *(const v4f*)(xr + k);
        v4f ev = *(const v4f*)(er + k);
        v4f q  = xv + (ev - xv);          // straight-through, matches x + (quantized - x)
        *(v4f*)(qr + k) = q;
        #pragma unroll
        for (int j = 0; j < 4; ++j) {
            const float dd = xv[j] - ev[j];
            msum += dd * dd;
            asum += fabsf(xv[j]);
        }
    }
    #pragma unroll
    for (int m = 1; m < 32; m <<= 1) {
        msum += __shfl_xor(msum, m, 32);
        asum += __shfl_xor(asum, m, 32);
    }
    __shared__ float bm[8], bp[8];
    if (lane == 0) {
        const float np = (asum > 0.f) ? 1.f : 0.f;
        bm[wave] = np * (msum * (1.0f / (float)D_DIM));
        bp[wave] = np;
        idxout[n] = (float)id;
        atomicAdd(&counts[id], 1u);
    }
    __syncthreads();
    if (threadIdx.x == 0) {
        float m = 0.f, p = 0.f;
        #pragma unroll
        for (int i = 0; i < 8; ++i) { m += bm[i]; p += bp[i]; }
        atomicAdd(loss_sum, m);
        atomicAdd(nonpad_sum, p);
    }
}

// ---------------- Kernel 4: perplexity + loss finalize (single block, M threads) ----------------
__global__ void vq_final_kernel(const unsigned int* __restrict__ counts,
                                const float* __restrict__ loss_sum,
                                const float* __restrict__ nonpad_sum,
                                float* __restrict__ loss_out, float* __restrict__ ppl_out,
                                float inv_n) {
    const int m = threadIdx.x;
    const float p = (float)counts[m] * inv_n;
    float h = p * __logf(p + 1e-10f);
    #pragma unroll
    for (int s = 1; s < 32; s <<= 1) h += __shfl_xor(h, s, 32);
    __shared__ float ps[32];
    if ((m & 31) == 0) ps[m >> 5] = h;
    __syncthreads();
    if (m < 32) {
        float t = ps[m];
        #pragma unroll
        for (int s = 1; s < 32; s <<= 1) t += __shfl_xor(t, s, 32);
        if (m == 0) {
            *ppl_out  = __expf(-t);
            *loss_out = 0.25f * loss_sum[0] / nonpad_sum[0];
        }
    }
}

extern "C" void kernel_launch(void* const* d_in, const int* in_sizes, int n_in,
                              void* d_out, int out_size, void* d_ws, size_t ws_size,
                              hipStream_t stream) {
    const float* x = (const float*)d_in[0];   // [N, 256]
    const float* e = (const float*)d_in[1];   // [M, 256]
    const int N = in_sizes[0] / D_DIM;        // 65536
    const int M = in_sizes[1] / D_DIM;        // 1024

    // Workspace layout (bytes):
    //   [0]   loss_sum (f32)   [4] nonpad_sum (f32)   [8..16) pad
    //   [16]  counts  u32[M]
    //   then  enorm   f32[M]
    //   then  idx     i32[N]
    //   then  eh      f16[M*256]
    float* ws_f           = (float*)d_ws;
    float* loss_sum       = ws_f + 0;
    float* nonpad_sum     = ws_f + 1;
    unsigned int* counts  = (unsigned int*)(ws_f + 4);
    float* enorm          = ws_f + 4 + M;
    int*   idx            = (int*)(ws_f + 4 + 2 * M);
    _Float16* eh          = (_Float16*)(ws_f + 4 + 2 * M + N);

    // Zero scalars + histogram (capturable memset node)
    hipMemsetAsync(d_ws, 0, (size_t)(16 + 4 * M), stream);

    // Output layout: quantized_st [N*256] | loss [1] | indices [N] | perplexity [1]
    float* out       = (float*)d_out;
    float* q_out     = out;
    float* loss_out  = out + (size_t)N * D_DIM;
    float* idx_out_f = out + (size_t)N * D_DIM + 1;
    float* ppl_out   = out + (size_t)N * D_DIM + 1 + N;

    vq_eprep_kernel<<<M, 256, 0, stream>>>(e, eh, enorm);
    vq_argmin_kernel<<<N / 128, 256, 0, stream>>>(x, eh, enorm, idx, M / 16);
    vq_out_kernel<<<N / 8, 256, 0, stream>>>(x, e, idx, q_out, idx_out_f,
                                             loss_sum, nonpad_sum, counts);
    vq_final_kernel<<<1, M, 0, stream>>>(counts, loss_sum, nonpad_sum,
                                         loss_out, ppl_out, 1.0f / (float)N);
}
